// CRFDecoder_86904368267595
// MI455X (gfx1250) — matrix operand order repaired and checked by reading, not compiled, not run on hardware
//
#include <hip/hip_runtime.h>
#include <hip/hip_bf16.h>

typedef __attribute__((ext_vector_type(16))) __bf16 v16bf;
typedef __attribute__((ext_vector_type(8)))  __bf16 v8bf;
typedef __attribute__((ext_vector_type(8)))  float  v8f;
typedef __attribute__((ext_vector_type(4)))  float  v4f;

#define TOKENS   32768      // B*T = 64*512
#define DIM      768
#define HID      384
#define LBL      9
#define LFULL    11         // LBL + start + end
#define SEQ_T    512
#define BATCH    64
#define NEGV     (-10000.0f)

static __device__ __forceinline__ v16bf cat16(v8bf lo, v8bf hi) {
    return __builtin_shufflevector(lo, hi, 0,1,2,3,4,5,6,7,8,9,10,11,12,13,14,15);
}
static __device__ __forceinline__ v8bf cvt8(v4f a, v4f b) {
    v8bf r;
    r[0] = (__bf16)a[0]; r[1] = (__bf16)a[1]; r[2] = (__bf16)a[2]; r[3] = (__bf16)a[3];
    r[4] = (__bf16)b[0]; r[5] = (__bf16)b[1]; r[6] = (__bf16)b[2]; r[7] = (__bf16)b[3];
    return r;
}

// ---------------------------------------------------------------------------
// K1: H = tanh(X @ W1^T + b1)   M=32768, N=384, K=768
// block = 256 threads = 8 waves; tile 128(M) x 64(N); K-chunk = 32 (bf16 WMMA)
// per wave: 16(M) x 64(N) -> 4 accumulators, 4 WMMAs / K-iter
// grid = (256, 6). Staging is vectorized (b128 loads, b128 LDS stores) and
// software-pipelined: next tile's global loads issue before this tile's WMMAs.
// ---------------------------------------------------------------------------
__global__ __launch_bounds__(256)
void crf_mlp1_wmma(const float* __restrict__ X, const float* __restrict__ W1,
                   const float* __restrict__ b1, __bf16* __restrict__ Hout) {
    __shared__ __bf16 As[128 * 32];   // [m][k]
    __shared__ __bf16 Bs[64 * 32];    // [n][k]  (n-major so K is contiguous)

    const int tid  = threadIdx.x;
    const int wave = tid >> 5;
    const int lane = tid & 31;
    const int hi   = lane >> 4;       // lane half-group
    const int ncol = lane & 15;
    const int m0   = blockIdx.x * 128;
    const int n0   = blockIdx.y * 64;

    // staging assignment: A: 16-float strip per thread; B: 8-float strip
    const int ar = tid >> 1, ac = (tid & 1) * 16;   // 128 rows x {0,16}
    const int br = tid >> 2, bc = (tid & 3) * 8;    // 64 rows  x {0,8,16,24}
    const float* aptr = X  + (size_t)(m0 + ar) * DIM + ac;
    const float* bptr = W1 + (size_t)(n0 + br) * DIM + bc;

    v4f fa[4], fb[2];
    #pragma unroll
    for (int i = 0; i < 4; ++i) fa[i] = *(const v4f*)(aptr + 4 * i);
    #pragma unroll
    for (int i = 0; i < 2; ++i) fb[i] = *(const v4f*)(bptr + 4 * i);

    v8f acc[4] = {v8f{}, v8f{}, v8f{}, v8f{}};

    for (int k0 = 0; k0 < DIM; k0 += 32) {
        __syncthreads();                       // prior frag reads done
        *(v8bf*)&As[ar * 32 + ac]     = cvt8(fa[0], fa[1]);
        *(v8bf*)&As[ar * 32 + ac + 8] = cvt8(fa[2], fa[3]);
        *(v8bf*)&Bs[br * 32 + bc]     = cvt8(fb[0], fb[1]);
        __syncthreads();

        // prefetch-issue next K-tile's global loads before the WMMAs
        if (k0 + 32 < DIM) {
            const float* ap = aptr + k0 + 32;
            const float* bp = bptr + k0 + 32;
            #pragma unroll
            for (int i = 0; i < 4; ++i) fa[i] = *(const v4f*)(ap + 4 * i);
            #pragma unroll
            for (int i = 0; i < 2; ++i) fb[i] = *(const v4f*)(bp + 4 * i);
            __builtin_prefetch(ap + 64, 0, 1);   // gfx1250 global_prefetch_b8
        }

        // fragments per CDNA5 16-bit layouts
        const int arow = wave * 16 + (lane & 15);
        v16bf afrag = cat16(*(const v8bf*)&As[arow * 32 + 8 * hi],
                            *(const v8bf*)&As[arow * 32 + 8 * (2 + hi)]);
        #pragma unroll
        for (int j = 0; j < 4; ++j) {
            int bo = (j * 16 + ncol) * 32 + 8 * hi;
            v16bf bfrag = cat16(*(const v8bf*)&Bs[bo], *(const v8bf*)&Bs[bo + 16]);
            acc[j] = __builtin_amdgcn_wmma_f32_16x16x32_bf16(
                         false, afrag, false, bfrag, (short)0, acc[j], false, false);
        }
    }

    // epilogue: C element r -> row = r + 8*hi, col = ncol
    const int mbase = m0 + wave * 16 + 8 * hi;
    #pragma unroll
    for (int j = 0; j < 4; ++j) {
        const float bias = b1[n0 + j * 16 + ncol];
        #pragma unroll
        for (int r = 0; r < 8; ++r) {
            Hout[(size_t)(mbase + r) * HID + n0 + j * 16 + ncol] =
                (__bf16)tanhf(acc[j][r] + bias);
        }
    }
}

// ---------------------------------------------------------------------------
// K2: logits = H @ W2^T + b2, padded to 16 cols (cols 9..15 = NEG)
// M=32768, N=16 (9 valid), K=384.  block = 256 threads = 8 waves (16 rows each)
// grid = 256
// ---------------------------------------------------------------------------
__global__ __launch_bounds__(256)
void crf_mlp2_wmma(const __bf16* __restrict__ Hin, const float* __restrict__ W2,
                   const float* __restrict__ b2, float* __restrict__ logits) {
    __shared__ __bf16 Ws[16 * HID];   // [n][k], n>=9 zero-filled

    const int tid  = threadIdx.x;
    const int wave = tid >> 5;
    const int lane = tid & 31;
    const int hi   = lane >> 4;
    const int ncol = lane & 15;
    const int m0   = blockIdx.x * 128;

    // stage W2 (zero-padded): 24-float contiguous strip per thread
    {
        const int n  = tid >> 4;          // 0..15
        const int kb = (tid & 15) * 24;   // 0..360
        if (n < LBL) {
            const float* wp = W2 + (size_t)n * HID + kb;
            v4f f[6];
            #pragma unroll
            for (int g = 0; g < 6; ++g) f[g] = *(const v4f*)(wp + 4 * g);
            *(v8bf*)&Ws[n * HID + kb]      = cvt8(f[0], f[1]);
            *(v8bf*)&Ws[n * HID + kb + 8]  = cvt8(f[2], f[3]);
            *(v8bf*)&Ws[n * HID + kb + 16] = cvt8(f[4], f[5]);
        } else {
            v8bf z = {};
            *(v8bf*)&Ws[n * HID + kb]      = z;
            *(v8bf*)&Ws[n * HID + kb + 8]  = z;
            *(v8bf*)&Ws[n * HID + kb + 16] = z;
        }
    }
    __syncthreads();

    const int row = m0 + wave * 16 + (lane & 15);
    v8f acc = {};
    #pragma unroll
    for (int k0 = 0; k0 < HID; k0 += 32) {
        v16bf afrag = cat16(*(const v8bf*)&Hin[(size_t)row * HID + k0 + 8 * hi],
                            *(const v8bf*)&Hin[(size_t)row * HID + k0 + 8 * (2 + hi)]);
        v16bf bfrag = cat16(*(const v8bf*)&Ws[ncol * HID + k0 + 8 * hi],
                            *(const v8bf*)&Ws[ncol * HID + k0 + 8 * (2 + hi)]);
        acc = __builtin_amdgcn_wmma_f32_16x16x32_bf16(
                  false, afrag, false, bfrag, (short)0, acc, false, false);
    }

    const float bias = (ncol < LBL) ? b2[ncol] : 0.0f;
    const int mbase = m0 + wave * 16 + 8 * hi;
    #pragma unroll
    for (int r = 0; r < 8; ++r) {
        float v = (ncol < LBL) ? (acc[r] + bias) : NEGV;
        logits[(size_t)(mbase + r) * 16 + ncol] = v;
    }
}

// ---------------------------------------------------------------------------
// K3: Viterbi forward + backtrack. One wave (32 threads) per batch element.
// lane = label (0..10 active; 9=start, 10=end)
// ---------------------------------------------------------------------------
__global__ __launch_bounds__(32)
void crf_viterbi(const float* __restrict__ logits, const int* __restrict__ mask,
                 const float* __restrict__ trans, int* __restrict__ preds) {
    __shared__ float vitS[16];
    __shared__ unsigned char bp[SEQ_T * 16];
    __shared__ int redI[32];

    const int b = blockIdx.x;
    const int l = threadIdx.x;

    // sequence length = sum(mask[b])
    int s = 0;
    for (int i = l; i < SEQ_T; i += 32) s += mask[b * SEQ_T + i];
    redI[l] = s;
    __syncthreads();
    if (l == 0) {
        int tot = 0;
        #pragma unroll
        for (int i = 0; i < 32; ++i) tot += redI[i];
        redI[0] = tot;
    }
    __syncthreads();
    const int len = redI[0];

    // per-lane transition row: trans[l][p] (score p -> l), end row value
    float trow[LFULL];
    float tEnd = 0.0f;
    if (l < LFULL) {
        #pragma unroll
        for (int p = 0; p < LFULL; ++p) trow[p] = trans[l * LFULL + p];
        tEnd = trans[(LFULL - 1) * LFULL + l];   // transition[end][l]
    }
    if (l < 16) vitS[l] = (l == LFULL - 2) ? 0.0f : NEGV;  // start label = 9
    __syncthreads();

    for (int t = 0; t < SEQ_T; ++t) {
        float logit = (l < LFULL) ? logits[((size_t)b * SEQ_T + t) * 16 + l] : NEGV;
        float best = NEGV;
        int   arg  = 0;
        float vcur = 0.0f;
        if (l < LFULL) {
            vcur = vitS[l];
            best = vitS[0] + trow[0];
            #pragma unroll
            for (int p = 1; p < LFULL; ++p) {
                float sc = vitS[p] + trow[p];
                if (sc > best) { best = sc; arg = p; }   // first-max (jnp.argmax)
            }
        }
        __syncthreads();
        if (l < LFULL) {
            float vnew = (t < len) ? (best + logit) : vcur;
            if (t == len - 1) vnew += tEnd;              // end transition at last step
            vitS[l] = vnew;
            bp[t * 16 + l] = (unsigned char)arg;
        }
        __syncthreads();
    }

    if (l == 0) {
        float best = vitS[0];
        int cur = 0;
        #pragma unroll
        for (int p = 1; p < LFULL; ++p)
            if (vitS[p] > best) { best = vitS[p]; cur = p; }
        preds[(size_t)b * SEQ_T + SEQ_T - 1] = cur;
        for (int t = SEQ_T - 1; t >= 1; --t) {
            cur = bp[t * 16 + cur];
            preds[(size_t)b * SEQ_T + t - 1] = cur;
        }
    }
}

// ---------------------------------------------------------------------------
extern "C" void kernel_launch(void* const* d_in, const int* in_sizes, int n_in,
                              void* d_out, int out_size, void* d_ws, size_t ws_size,
                              hipStream_t stream) {
    const float* X     = (const float*)d_in[0];   // [64,512,768]
    const int*   maskp = (const int*)  d_in[1];   // [64,512]
    const float* W1    = (const float*)d_in[2];   // [384,768]
    const float* b1    = (const float*)d_in[3];   // [384]
    const float* W2    = (const float*)d_in[4];   // [9,384]
    const float* b2    = (const float*)d_in[5];   // [9]
    const float* trans = (const float*)d_in[6];   // [11,11]
    int* preds = (int*)d_out;                     // [64,512] int32 labels

    __bf16* Hbuf   = (__bf16*)d_ws;                                       // 24 MB
    float*  logitb = (float*)((char*)d_ws + (size_t)TOKENS * HID * 2);    //  2 MB

    dim3 g1(TOKENS / 128, HID / 64);
    crf_mlp1_wmma<<<g1, 256, 0, stream>>>(X, W1, b1, Hbuf);
    crf_mlp2_wmma<<<TOKENS / 128, 256, 0, stream>>>(Hbuf, W2, b2, logitb);
    crf_viterbi<<<BATCH, 32, 0, stream>>>(logitb, maskp, trans, preds);
}
